// RBF_model_19705309954372
// MI455X (gfx1250) — matrix operand order repaired and checked by reading, not compiled, not run on hardware
//
#include <hip/hip_runtime.h>
#include <math.h>

typedef _Float16 v16h __attribute__((ext_vector_type(16)));
typedef _Float16 v8h  __attribute__((ext_vector_type(8)));
typedef _Float16 v4h  __attribute__((ext_vector_type(4)));
typedef float    v8f  __attribute__((ext_vector_type(8)));

#define B_DIM   2048
#define F_DIM   512
#define C_DIM   100
#define M_TILE  64      // batch rows per block
#define N_CHUNK 64      // m-columns processed per chunk (2 x 16 cols per wave)
#define K_STEP  32      // WMMA K per instruction
#define LDW     528     // padded LDS row stride in halves (1056 B = 8 banks skew)

// distances = mean_F((z-c)^2) / (2*sigma^2)  ->  scale = 1/(F * 2 * 0.1^2)
#define DIST_SCALE (1.0f / (512.0f * 0.02f))

__global__ __launch_bounds__(256)
void rbf_wmma_kernel(const float* __restrict__ x,
                     const float* __restrict__ kernels,
                     const float* __restrict__ mvals,
                     const float* __restrict__ Nvals,
                     float* __restrict__ out)
{
    extern __shared__ __align__(16) char smem[];
    _Float16* As     = (_Float16*)smem;              // [M_TILE][LDW]  x tile (f16)
    _Float16* Bs     = As + M_TILE * LDW;            // [N_CHUNK][LDW] kernel tile (f16)
    float*    rowsum = (float*)(Bs + N_CHUNK * LDW); // [M_TILE]

    const int tid  = threadIdx.x;
    const int lane = tid & 31;
    const int wave = tid >> 5;       // 0..7
    const int wr   = wave >> 1;      // 0..3 : 16-row slab within tile
    const int wc   = wave & 1;       // 0..1 : 32-col slab within chunk (2 tiles)
    const int i0   = blockIdx.x * M_TILE;   // batch row base
    const int n    = blockIdx.y;            // class index

    if (tid < M_TILE) rowsum[tid] = 0.0f;

    // ---- stage x tile [M_TILE x F_DIM] f32 -> f16 into LDS (once) ----
    {
        const float4* xg = (const float4*)(x + (size_t)i0 * F_DIM);
        for (int idx = tid; idx < M_TILE * (F_DIM / 4); idx += 256) {
            const int row = idx >> 7;        // / (F_DIM/4)
            const int c4  = idx & 127;
            const float4 v = xg[(size_t)row * (F_DIM / 4) + c4];
            v4h h = { (_Float16)v.x, (_Float16)v.y, (_Float16)v.z, (_Float16)v.w };
            *(v4h*)(As + row * LDW + c4 * 4) = h;
        }
    }

    const float invN = 1.0f / Nvals[n];

    float acc0[8], acc1[8];
    #pragma unroll
    for (int q = 0; q < 8; ++q) { acc0[q] = 0.0f; acc1[q] = 0.0f; }

    // Per-lane LDS fragment bases (contiguous k order, k = 16*half + s, same for A and B):
    //  A 16x32 f16: lane half selects K-offset 0 / 16; elems 0-7 = K j0..j0+7, 8-15 = K j0+8..j0+15
    //  B 32x16 f16: N = lane&15, K contiguous, lane half selects K-offset 0 / 16
    const _Float16* aBase  = As + (wr * 16 + (lane & 15)) * LDW + (lane >> 4) * 16;
    const _Float16* bBase0 = Bs + (wc * 32 + (lane & 15)) * LDW + (lane >> 4) * 16;
    const _Float16* bBase1 = bBase0 + 16 * LDW;   // second 16-col tile

    for (int mc = 0; mc < F_DIM; mc += N_CHUNK) {
        __syncthreads();  // previous chunk's readers done before overwriting Bs

        // ---- stage kernels[mc..mc+63, n, :] f32 -> f16 into LDS ----
        {
            const float* kg = kernels + (size_t)mc * (C_DIM * F_DIM) + (size_t)n * F_DIM;
            for (int idx = tid; idx < N_CHUNK * (F_DIM / 4); idx += 256) {
                const int row = idx >> 7;
                const int c4  = idx & 127;
                const float4 v = *(const float4*)(kg + (size_t)row * (C_DIM * F_DIM) + c4 * 4);
                v4h h = { (_Float16)v.x, (_Float16)v.y, (_Float16)v.z, (_Float16)v.w };
                *(v4h*)(Bs + row * LDW + c4 * 4) = h;
            }
        }
        __syncthreads();

        // ---- GEMM: two 16x16 tiles per wave sharing the A fragment ----
        v8f cfrag0 = {0.0f, 0.0f, 0.0f, 0.0f, 0.0f, 0.0f, 0.0f, 0.0f};
        v8f cfrag1 = {0.0f, 0.0f, 0.0f, 0.0f, 0.0f, 0.0f, 0.0f, 0.0f};
        #pragma unroll
        for (int k0 = 0; k0 < F_DIM; k0 += K_STEP) {
            union { v16h v; v8h h[2]; } a, b0, b1;
            a.h[0]  = *(const v8h*)(aBase + k0);        // K j0   .. j0+7
            a.h[1]  = *(const v8h*)(aBase + k0 + 8);    // K j0+8 .. j0+15
            b0.h[0] = *(const v8h*)(bBase0 + k0);
            b0.h[1] = *(const v8h*)(bBase0 + k0 + 8);
            b1.h[0] = *(const v8h*)(bBase1 + k0);
            b1.h[1] = *(const v8h*)(bBase1 + k0 + 8);
            cfrag0 = __builtin_amdgcn_wmma_f32_16x16x32_f16(
                        false, a.v, false, b0.v, (short)0, cfrag0, false, false);
            cfrag1 = __builtin_amdgcn_wmma_f32_16x16x32_f16(
                        false, a.v, false, b1.v, (short)0, cfrag1, false, false);
        }

        // ---- fused epilogue: acc += (z - m/N)^2 for this m-chunk ----
        // C layout: cfrag[q] -> row = wr*16 + q + 8*(lane>>4), col m = mc + wc*32 + (lane&15) (+16)
        const int   mcol0 = mc + wc * 32 + (lane & 15);
        const float cen0  = mvals[(size_t)mcol0 * C_DIM + n] * invN;
        const float cen1  = mvals[(size_t)(mcol0 + 16) * C_DIM + n] * invN;
        #pragma unroll
        for (int q = 0; q < 8; ++q) {
            const float d0 = cfrag0[q] - cen0;
            const float d1 = cfrag1[q] - cen1;
            acc0[q] += d0 * d0;
            acc1[q] += d1 * d1;
        }
    }

    // ---- reduce over the 16 lanes sharing each output row, then over waves ----
    // Both tiles of a wave map to the same rows -> merge accumulators first.
    #pragma unroll
    for (int q = 0; q < 8; ++q) {
        float v = acc0[q] + acc1[q];
        v += __shfl_xor(v, 1, 16);
        v += __shfl_xor(v, 2, 16);
        v += __shfl_xor(v, 4, 16);
        v += __shfl_xor(v, 8, 16);
        if ((lane & 15) == 0) {
            const int row = wr * 16 + q + ((lane >> 4) << 3);
            atomicAdd(&rowsum[row], v);
        }
    }
    __syncthreads();

    if (tid < M_TILE) {
        const float dist = rowsum[tid] * DIST_SCALE;
        out[(size_t)(i0 + tid) * C_DIM + n] = __expf(-dist);
    }
}

extern "C" void kernel_launch(void* const* d_in, const int* in_sizes, int n_in,
                              void* d_out, int out_size, void* d_ws, size_t ws_size,
                              hipStream_t stream) {
    (void)in_sizes; (void)n_in; (void)d_ws; (void)ws_size; (void)out_size;
    const float* x       = (const float*)d_in[0];
    const float* kernels = (const float*)d_in[1];
    const float* mvals   = (const float*)d_in[2];
    const float* Nvals   = (const float*)d_in[3];
    float*       out     = (float*)d_out;

    const int smem_bytes = (M_TILE + N_CHUNK) * LDW * (int)sizeof(_Float16)
                         + M_TILE * (int)sizeof(float);   // ~135.4 KB
    (void)hipFuncSetAttribute((const void*)rbf_wmma_kernel,
                              hipFuncAttributeMaxDynamicSharedMemorySize, smem_bytes);

    dim3 grid(B_DIM / M_TILE, C_DIM);   // (32, 100)
    dim3 block(256);                    // 8 waves (wave32)
    rbf_wmma_kernel<<<grid, block, smem_bytes, stream>>>(x, kernels, mvals, Nvals, out);
}
